// GPTModel_31714038514328
// MI455X (gfx1250) — matrix operand order repaired and checked by reading, not compiled, not run on hardware
//
#include <hip/hip_runtime.h>
#include <hip/hip_bf16.h>

typedef __attribute__((ext_vector_type(16))) _Float16 v16h;
typedef __attribute__((ext_vector_type(8)))  _Float16 half8;
typedef __attribute__((ext_vector_type(8)))  float    v8f;

#define D_MODEL 768
#define N_HEADS 12
#define HEAD_DIM 64
#define SEQ 1024
#define BATCH 2
#define M_TOK (BATCH*SEQ)
#define FFDIM 3072
#define NLAYERS 4
#define VOCAB 50257

__device__ __forceinline__ float gelu_f(float x) {
    const float c = 0.7978845608028654f; // sqrt(2/pi)
    float t = tanhf(c * (x + 0.044715f * x * x * x));
    return 0.5f * x * (1.0f + t);
}

// x[m,d] = tok_emb[idx[m]][d] + pos_emb[m % SEQ][d]
__global__ void embed_kernel(const int* __restrict__ idx,
                             const float* __restrict__ tok,
                             const float* __restrict__ pos,
                             float* __restrict__ x) {
    int i = blockIdx.x * blockDim.x + threadIdx.x;
    if (i >= M_TOK * D_MODEL) return;
    int m = i / D_MODEL, d = i - m * D_MODEL;
    int t = m & (SEQ - 1);
    x[i] = tok[(size_t)idx[m] * D_MODEL + d] + pos[(size_t)t * D_MODEL + d];
}

// per-row layernorm, fp32 in -> f16 out
__global__ __launch_bounds__(256) void layernorm_f16(
    const float* __restrict__ X, const float* __restrict__ scale,
    const float* __restrict__ shift, _Float16* __restrict__ out) {
    __shared__ float red[256];
    const int row = blockIdx.x;
    const int tid = threadIdx.x;
    const float* x = X + (size_t)row * D_MODEL;
    float v[3];
    float s = 0.f;
#pragma unroll
    for (int i = 0; i < 3; i++) { v[i] = x[tid + i * 256]; s += v[i]; }
    red[tid] = s; __syncthreads();
    for (int o = 128; o > 0; o >>= 1) { if (tid < o) red[tid] += red[tid + o]; __syncthreads(); }
    float mean = red[0] * (1.f / 768.f);
    __syncthreads();
    s = 0.f;
#pragma unroll
    for (int i = 0; i < 3; i++) { float d = v[i] - mean; s += d * d; }
    red[tid] = s; __syncthreads();
    for (int o = 128; o > 0; o >>= 1) { if (tid < o) red[tid] += red[tid + o]; __syncthreads(); }
    float rstd = rsqrtf(red[0] * (1.f / 768.f) + 1e-5f);
#pragma unroll
    for (int i = 0; i < 3; i++) {
        int d = tid + i * 256;
        out[(size_t)row * D_MODEL + d] = (_Float16)(scale[d] * (v[i] - mean) * rstd + shift[d]);
    }
}

// Generic WMMA GEMM: C[M,N] = A(f16)[M,K] @ B(f32->f16)[K,N]  (+bias/epilogue)
// Block tile 128x128, 8 waves (4 M x 2 N), wave tile 32x64, K-step 32.
// modes: 0 fp32 out; 1 fp32 resid-add; 2 f16 GELU out; 3 f16 [b,h,n,hd]; 4 f16 [b,h,hd,n]
__global__ __launch_bounds__(256) void gemm_wmma(
    const _Float16* __restrict__ A, const float* __restrict__ B,
    const float* __restrict__ bias, const float* __restrict__ resid,
    float* __restrict__ outF, _Float16* __restrict__ outH,
    int Mtot, int Ntot, int Ktot, int mode) {
    __shared__ __align__(16) _Float16 As[128 * 32];   // [m][k]
    __shared__ __align__(16) _Float16 Bs[128 * 40];   // [n][k], padded stride 40
    const int tid  = threadIdx.x;
    const int lane = tid & 31;
    const int wave = tid >> 5;
    const int wm   = wave & 3;   // 0..3  (M direction, 32 rows each)
    const int wn   = wave >> 2;  // 0..1  (N direction, 64 cols each)
    const int nn   = lane & 15;
    const int hf   = lane >> 4;
    const int m0   = blockIdx.y * 128;
    const int n0   = blockIdx.x * 128;
    (void)Mtot;

    const v8f vz = {0.f, 0.f, 0.f, 0.f, 0.f, 0.f, 0.f, 0.f};
    v8f acc[2][4];
    for (int i = 0; i < 2; i++) for (int j = 0; j < 4; j++) acc[i][j] = vz;

    // --- hoisted staging addresses ---
    const int arow = tid >> 1;          // 0..127
    const int aseg = (tid & 1) * 16;    // 0 or 16
    const _Float16* aptr = A + (size_t)(m0 + arow) * Ktot + aseg;
    _Float16* asl = &As[arow * 32 + aseg];

    const int t7   = tid >> 7;          // 0..1  (k row parity)
    const int ncol = tid & 127;         // column within tile, j/k0 invariant
    const int colg = n0 + ncol;
    const int cl   = colg < Ntot ? colg : Ntot - 1;   // branchless clamp
    // GVS-form addressing: uniform per-j SGPR pointer + one shared 32-bit lane offset
    const float* bbase = B;                            // advances 32*Ntot per K-step
    const int voff  = cl + t7 * Ntot;                  // per-lane, loop invariant (<2^31)
    const int jstep = 2 * Ntot;                        // uniform per-j stride
    _Float16* bsl = &Bs[ncol * 40 + t7];

    for (int k0 = 0; k0 < Ktot; k0 += 32) {
        __syncthreads();
        // A tile: issue both b128 loads first; latency hidden behind the B loads
        const half8* ga = reinterpret_cast<const half8*>(aptr + k0);
        half8 a0 = ga[0];
        half8 a1 = ga[1];
        // B tile: 16 independent loads, uniform SGPR base per j + shared VGPR offset
        float tmp[16];
#pragma unroll
        for (int j = 0; j < 16; j++) {
            const float* bj = bbase + j * jstep;       // uniform (SALU) pointer
            tmp[j] = bj[voff];
        }
        // drain into LDS
        half8* la = reinterpret_cast<half8*>(asl);
        la[0] = a0; la[1] = a1;
#pragma unroll
        for (int j = 0; j < 16; j++) bsl[2 * j] = (_Float16)tmp[j];
        bbase += (size_t)32 * Ntot;
        // prefetch next K-step tiles (global_prefetch_b8)
        if (k0 + 32 < Ktot) {
            __builtin_prefetch(aptr + k0 + 32, 0, 3);
            __builtin_prefetch(bbase + voff, 0, 3);
        }
        __syncthreads();

        v16h afrag[2], bfrag[4];
#pragma unroll
        for (int mi = 0; mi < 2; mi++) {
            int r = wm * 32 + mi * 16 + nn;
            union { v16h v; half8 h8[2]; } u;
            u.h8[0] = *reinterpret_cast<const half8*>(&As[r * 32 + hf * 8]);
            u.h8[1] = *reinterpret_cast<const half8*>(&As[r * 32 + 16 + hf * 8]);
            afrag[mi] = u.v;
        }
#pragma unroll
        for (int ni = 0; ni < 4; ni++) {
            int c = wn * 64 + ni * 16 + nn;
            union { v16h v; half8 h8[2]; } u;
            u.h8[0] = *reinterpret_cast<const half8*>(&Bs[c * 40 + hf * 8]);
            u.h8[1] = *reinterpret_cast<const half8*>(&Bs[c * 40 + hf * 8 + 16]);
            bfrag[ni] = u.v;
        }
#pragma unroll
        for (int mi = 0; mi < 2; mi++)
#pragma unroll
            for (int ni = 0; ni < 4; ni++)
                acc[mi][ni] = __builtin_amdgcn_wmma_f32_16x16x32_f16(
                    false, afrag[mi], false, bfrag[ni], (short)0, acc[mi][ni], false, false);
    }

    // epilogue: C/D layout -> lane nn is column, VGPR r is row r+8*hf.
    // All indices advance incrementally (no per-element 64-bit multiplies).
#pragma unroll
    for (int mi = 0; mi < 2; mi++) {
        const int row0 = m0 + wm * 32 + mi * 16 + 8 * hf;   // 8 consecutive rows from here
        const int b  = row0 >> 10;           // batch (row block never crosses 1024)
        const int t0 = row0 & (SEQ - 1);
#pragma unroll
        for (int ni = 0; ni < 4; ni++) {
            const int col = n0 + wn * 64 + ni * 16 + nn;
            if (col >= Ntot) continue;
            const float bv = bias ? bias[col] : 0.f;
            size_t idx = (size_t)row0 * Ntot + col;                                     // step Ntot
            const int hcol = col >> 6, hd = col & 63;
            size_t idx3 = (((size_t)(b * N_HEADS + hcol)) * SEQ + t0) * HEAD_DIM + hd;  // step HEAD_DIM
            size_t idx4 = (((size_t)(b * N_HEADS + hcol)) * HEAD_DIM + hd) * SEQ + t0;  // step 1
#pragma unroll
            for (int r = 0; r < 8; r++) {
                float v = acc[mi][ni][r] + bv;
                if (mode == 0) {
                    outF[idx] = v;
                } else if (mode == 1) {
                    outF[idx] = resid[idx] + v;
                } else if (mode == 2) {
                    outH[idx] = (_Float16)gelu_f(v);
                } else if (mode == 3) {
                    outH[idx3] = (_Float16)v;
                } else {
                    outH[idx4] = (_Float16)v;
                }
                idx += Ntot; idx3 += HEAD_DIM; idx4 += 1;
            }
        }
    }
}

// Flash-style causal attention. Q,K: f16 [b,h,n,hd]; VT: f16 [b,h,hd,n]; ctx: f16 [b,n,D]
// block = 128 threads = 4 waves; each wave owns 16 query rows.
__global__ __launch_bounds__(128) void attn_wmma(
    const _Float16* __restrict__ Q, const _Float16* __restrict__ K,
    const _Float16* __restrict__ VT, const int* __restrict__ amask,
    _Float16* __restrict__ ctx) {
    __shared__ __align__(16) _Float16 P[4 * 16 * 32];
    const int lane = threadIdx.x & 31;
    const int wave = threadIdx.x >> 5;
    const int nn = lane & 15;
    const int hf = lane >> 4;
    const int bh = blockIdx.x >> 4;   // b*H + h
    const int qt = blockIdx.x & 15;
    const int b = bh / N_HEADS;
    const int h = bh - b * N_HEADS;
    const int qbase = qt * 64 + wave * 16;
    const float scale = 0.125f;       // 1/sqrt(64)

    // Q fragments (16 rows x 64 dims = two 16x32 A-frags)
    v16h aq[2];
#pragma unroll
    for (int c = 0; c < 2; c++) {
        const _Float16* qp = Q + ((size_t)bh * SEQ + qbase + nn) * HEAD_DIM + c * 32;
        union { v16h v; half8 h8[2]; } u;
        u.h8[0] = *reinterpret_cast<const half8*>(qp + hf * 8);
        u.h8[1] = *reinterpret_cast<const half8*>(qp + 16 + hf * 8);
        aq[c] = u.v;
    }

    const v8f vz = {0.f, 0.f, 0.f, 0.f, 0.f, 0.f, 0.f, 0.f};
    float m[8], l[8];
    v8f accO[4];
#pragma unroll
    for (int r = 0; r < 8; r++) { m[r] = -1e30f; l[r] = 0.f; }
    for (int g = 0; g < 4; g++) accO[g] = vz;

    _Float16* pw = &P[wave * 16 * 32];
    const int nkb = ((qbase + 15) >> 5) + 1;

    for (int kb = 0; kb < nkb; kb++) {
        // S = Q @ K^T for 32 keys (two 16-key groups, K=64 over two chunks)
        v8f s[2];
#pragma unroll
        for (int g = 0; g < 2; g++) {
            s[g] = vz;
#pragma unroll
            for (int c = 0; c < 2; c++) {
                int key = kb * 32 + g * 16 + nn;
                const _Float16* kp = K + ((size_t)bh * SEQ + key) * HEAD_DIM + c * 32 + hf * 8;
                union { v16h v; half8 h8[2]; } u;
                u.h8[0] = *reinterpret_cast<const half8*>(kp);
                u.h8[1] = *reinterpret_cast<const half8*>(kp + 16);
                s[g] = __builtin_amdgcn_wmma_f32_16x16x32_f16(
                    false, aq[c], false, u.v, (short)0, s[g], false, false);
            }
        }
        // masking + online softmax over the 32-key block
        float sv[2][8];
        float newm[8];
#pragma unroll
        for (int r = 0; r < 8; r++) newm[r] = m[r];
#pragma unroll
        for (int g = 0; g < 2; g++) {
            int col = kb * 32 + g * 16 + nn;
            int ok = amask[b * SEQ + col];
#pragma unroll
            for (int r = 0; r < 8; r++) {
                int row = qbase + r + 8 * hf;
                float v = s[g][r] * scale;
                if (col > row || ok == 0) v = -1e30f;
                sv[g][r] = v;
                float rm = v;
                for (int off = 1; off < 16; off <<= 1) rm = fmaxf(rm, __shfl_xor(rm, off, 32));
                newm[r] = fmaxf(newm[r], rm);
            }
        }
#pragma unroll
        for (int r = 0; r < 8; r++) {
            float corr = __expf(m[r] - newm[r]);
            float psum = 0.f;
#pragma unroll
            for (int g = 0; g < 2; g++) {
                float p = __expf(sv[g][r] - newm[r]);
                pw[(r + 8 * hf) * 32 + g * 16 + nn] = (_Float16)p;  // C-layout -> [row][key] in LDS
                float rs = p;
                for (int off = 1; off < 16; off <<= 1) rs += __shfl_xor(rs, off, 32);
                psum += rs;
            }
            l[r] = l[r] * corr + psum;
            m[r] = newm[r];
#pragma unroll
            for (int g2 = 0; g2 < 4; g2++) accO[g2][r] *= corr;
        }
        // reload P as a 16x32 A-fragment (wave-private LDS; per-wave DS ordering)
        union { v16h v; half8 h8[2]; } up;
        up.h8[0] = *reinterpret_cast<const half8*>(&pw[nn * 32 + hf * 8]);
        up.h8[1] = *reinterpret_cast<const half8*>(&pw[nn * 32 + 16 + hf * 8]);
        v16h pa = up.v;
        // O += P @ V  (V transposed: keys contiguous per dim)
#pragma unroll
        for (int g2 = 0; g2 < 4; g2++) {
            int dim = g2 * 16 + nn;
            const _Float16* vp = VT + ((size_t)bh * HEAD_DIM + dim) * SEQ + kb * 32 + hf * 8;
            union { v16h v; half8 h8[2]; } u;
            u.h8[0] = *reinterpret_cast<const half8*>(vp);
            u.h8[1] = *reinterpret_cast<const half8*>(vp + 16);
            accO[g2] = __builtin_amdgcn_wmma_f32_16x16x32_f16(
                false, pa, false, u.v, (short)0, accO[g2], false, false);
        }
    }
    // normalize and store ctx in [b, n, D] for the projection GEMM
#pragma unroll
    for (int r = 0; r < 8; r++) {
        int row = qbase + r + 8 * hf;
        float inv = 1.f / fmaxf(l[r], 1e-20f);
#pragma unroll
        for (int g2 = 0; g2 < 4; g2++) {
            float v = accO[g2][r] * inv;
            ctx[((size_t)b * SEQ + row) * D_MODEL + h * HEAD_DIM + g2 * 16 + nn] = (_Float16)v;
        }
    }
}

extern "C" void kernel_launch(void* const* d_in, const int* in_sizes, int n_in,
                              void* d_out, int out_size, void* d_ws, size_t ws_size,
                              hipStream_t stream) {
    (void)in_sizes; (void)n_in; (void)out_size; (void)ws_size;
    const int*   in_idx = (const int*)d_in[0];
    const int*   amask  = (const int*)d_in[1];
    const float* tok    = (const float*)d_in[2];
    const float* pos    = (const float*)d_in[3];
    const float* wq     = (const float*)d_in[4];
    const float* wk     = (const float*)d_in[5];
    const float* wv     = (const float*)d_in[6];
    const float* wo     = (const float*)d_in[7];
    const float* bo     = (const float*)d_in[8];
    const float* n1s    = (const float*)d_in[9];
    const float* n1b    = (const float*)d_in[10];
    const float* n2s    = (const float*)d_in[11];
    const float* n2b    = (const float*)d_in[12];
    const float* w1     = (const float*)d_in[13];
    const float* b1     = (const float*)d_in[14];
    const float* w2     = (const float*)d_in[15];
    const float* b2     = (const float*)d_in[16];
    const float* fns    = (const float*)d_in[17];
    const float* fnb    = (const float*)d_in[18];
    const float* whead  = (const float*)d_in[19];
    float* out = (float*)d_out;

    char* ws = (char*)d_ws;
    float*    x    = (float*)ws;    ws += (size_t)M_TOK * D_MODEL * 4;
    _Float16* ln16 = (_Float16*)ws; ws += (size_t)M_TOK * D_MODEL * 2;
    _Float16* ff16 = (_Float16*)ws; ws += (size_t)M_TOK * FFDIM * 2;
    _Float16* q16  = (_Float16*)ws; ws += (size_t)M_TOK * D_MODEL * 2;
    _Float16* k16  = (_Float16*)ws; ws += (size_t)M_TOK * D_MODEL * 2;
    _Float16* vT16 = (_Float16*)ws; ws += (size_t)M_TOK * D_MODEL * 2;
    _Float16* c16  = (_Float16*)ws; ws += (size_t)M_TOK * D_MODEL * 2;

    dim3 blk(256);
    embed_kernel<<<dim3((M_TOK * D_MODEL + 255) / 256), blk, 0, stream>>>(in_idx, tok, pos, x);

    auto gemm = [&](const _Float16* A, const float* B, const float* bias,
                    const float* resid, float* oF, _Float16* oH,
                    int M, int N, int Kd, int mode) {
        dim3 g((N + 127) / 128, M / 128);
        gemm_wmma<<<g, dim3(256), 0, stream>>>(A, B, bias, resid, oF, oH, M, N, Kd, mode);
    };

    for (int l = 0; l < NLAYERS; l++) {
        layernorm_f16<<<dim3(M_TOK), blk, 0, stream>>>(x, n1s + l * D_MODEL, n1b + l * D_MODEL, ln16);
        gemm(ln16, wq + (size_t)l * D_MODEL * D_MODEL, nullptr, nullptr, nullptr, q16,
             M_TOK, D_MODEL, D_MODEL, 3);
        gemm(ln16, wk + (size_t)l * D_MODEL * D_MODEL, nullptr, nullptr, nullptr, k16,
             M_TOK, D_MODEL, D_MODEL, 3);
        gemm(ln16, wv + (size_t)l * D_MODEL * D_MODEL, nullptr, nullptr, nullptr, vT16,
             M_TOK, D_MODEL, D_MODEL, 4);
        attn_wmma<<<dim3(BATCH * N_HEADS * (SEQ / 64)), dim3(128), 0, stream>>>(
            q16, k16, vT16, amask, c16);
        gemm(c16, wo + (size_t)l * D_MODEL * D_MODEL, bo + l * D_MODEL, x, x, nullptr,
             M_TOK, D_MODEL, D_MODEL, 1);
        layernorm_f16<<<dim3(M_TOK), blk, 0, stream>>>(x, n2s + l * D_MODEL, n2b + l * D_MODEL, ln16);
        gemm(ln16, w1 + (size_t)l * D_MODEL * FFDIM, b1 + l * FFDIM, nullptr, nullptr, ff16,
             M_TOK, FFDIM, D_MODEL, 2);
        gemm(ff16, w2 + (size_t)l * FFDIM * D_MODEL, b2 + l * D_MODEL, x, x, nullptr,
             M_TOK, D_MODEL, FFDIM, 1);
    }
    layernorm_f16<<<dim3(M_TOK), blk, 0, stream>>>(x, fns, fnb, ln16);
    gemm(ln16, whead, nullptr, nullptr, out, nullptr, M_TOK, VOCAB, D_MODEL, 0);
}
